// dummylearner_30056181138026
// MI455X (gfx1250) — matrix operand order repaired and checked by reading, not compiled, not run on hardware
//
#include <hip/hip_runtime.h>
#include <hip/hip_bf16.h>

// ---------------------------------------------------------------------------
// LSTM (INPUT=100, HIDDEN=6, T=2048, B=256) for gfx1250 (MI455X).
// Phase 1: xg = x @ W_ih^T + (b_ih+b_hh)   -- WMMA f16->f32, HBM-bound
// Phase 2: sequential recurrence, one batch per LANE-QUAD (gate group per
//          lane), shared via ds_swizzle group-of-4 broadcast (wave32).
// ---------------------------------------------------------------------------

#define T_STEPS 2048
#define BATCH   256
#define IN_F    100
#define NH      24                 // 4*HIDDEN
#define HID     6
#define M_TOT   (T_STEPS * BATCH)  // 524288 rows
#define M_TILES (M_TOT / 16)       // 32768 16-row tiles

typedef __attribute__((ext_vector_type(16))) _Float16 v16h;
typedef __attribute__((ext_vector_type(8)))  float    v8f;

// ---------------------------------------------------------------------------
// Phase 1: input projection GEMM via v_wmma_f32_16x16x32_f16.
//   M = 524288 (t*B+b), K = 100 -> padded 128 (4 chunks of 32),
//   N = 24 -> padded 32 (2 tiles of 16).
// Each wave keeps the full padded B operand (W_ih^T) in 64 VGPRs and streams
// M-tiles; 8 WMMAs per tile.  All A-loads are branch-free (tail chunk is an
// unconditional in-bounds load + v_cndmask), so the hot loop has uniform flow.
// ---------------------------------------------------------------------------
__device__ __forceinline__ void cvt8(const float4 q0, const float4 q1,
                                     _Float16* dst) {
  dst[0] = (_Float16)q0.x; dst[1] = (_Float16)q0.y;
  dst[2] = (_Float16)q0.z; dst[3] = (_Float16)q0.w;
  dst[4] = (_Float16)q1.x; dst[5] = (_Float16)q1.y;
  dst[6] = (_Float16)q1.z; dst[7] = (_Float16)q1.w;
}

__global__ __launch_bounds__(256) void lstm_xproj_wmma(
    const float* __restrict__ x, const float* __restrict__ W_ih,
    const float* __restrict__ b_ih, const float* __restrict__ b_hh,
    float* __restrict__ xg) {
  const int lane  = threadIdx.x & 31;
  const int grp   = lane >> 4;     // which 16-lane half
  const int lo    = lane & 15;
  const int gwave = blockIdx.x * 8 + (threadIdx.x >> 5);
  const int nWaves = gridDim.x * 8;

  // B operand: B[k][n] = W_ih[n][k], zero-padded.  CDNA5 16-bit B layout:
  // N = lane&15; lanes 0-15 hold chunk K {0..7,16..23}, lanes 16-31 K {8..15,24..31}.
  v16h Bt[2][4];
#pragma unroll
  for (int nt = 0; nt < 2; ++nt) {
    const int n = nt * 16 + lo;
#pragma unroll
    for (int kc = 0; kc < 4; ++kc) {
      v16h bv;
#pragma unroll
      for (int e = 0; e < 16; ++e) {
        const int k = kc * 32 + grp * 8 + (e < 8 ? e : e + 8);
        const float v = (n < NH && k < IN_F) ? W_ih[n * IN_F + k] : 0.f;
        bv[e] = (_Float16)v;
      }
      Bt[nt][kc] = bv;
    }
  }
  const float bias0 = b_ih[lo] + b_hh[lo];
  const float bias1 = (lo < 8) ? (b_ih[16 + lo] + b_hh[16 + lo]) : 0.f;
  const bool tailLane = (grp == 0);   // only K-low half holds K=96..99

  for (int tile = gwave; tile < M_TILES; tile += nWaves) {
    const int m0 = tile * 16;
    // CDNA5 16-bit A layout: M = lane&15; lane half selects K sub-blocks
    // {0..7,16..23} (lanes 0-15) vs {8..15,24..31} (lanes 16-31) per chunk.
    const float* row = x + (size_t)(m0 + lo) * IN_F;
    v8f acc0 = {};
    v8f acc1 = {};
    v16h a[4];
#pragma unroll
    for (int kc = 0; kc < 3; ++kc) {    // full chunks, max index 88+8=96<=100
      _Float16 tmp[16];
      const int b0 = kc * 32 + grp * 8;
      cvt8(*(const float4*)(row + b0),      *(const float4*)(row + b0 + 4),  tmp);
      cvt8(*(const float4*)(row + b0 + 16), *(const float4*)(row + b0 + 20), tmp + 8);
#pragma unroll
      for (int e = 0; e < 16; ++e) a[kc][e] = tmp[e];
    }
    {   // tail chunk kc=3: only grp==0 lanes, elements 0..3 (K=96..99) valid
      const float4 q = *(const float4*)(row + 96);   // in-bounds for all lanes
      v16h av = {};
      av[0] = (_Float16)(tailLane ? q.x : 0.f);
      av[1] = (_Float16)(tailLane ? q.y : 0.f);
      av[2] = (_Float16)(tailLane ? q.z : 0.f);
      av[3] = (_Float16)(tailLane ? q.w : 0.f);
      a[3] = av;
    }
#pragma unroll
    for (int kc = 0; kc < 4; ++kc) {
      acc0 = __builtin_amdgcn_wmma_f32_16x16x32_f16(false, a[kc], false, Bt[0][kc],
                                                    (short)0, acc0, false, false);
      acc1 = __builtin_amdgcn_wmma_f32_16x16x32_f16(false, a[kc], false, Bt[1][kc],
                                                    (short)0, acc1, false, false);
    }
    // C/D layout: VGPR r, lanes 0-15 -> (M=m0+r, N=lo); lanes 16-31 -> M=m0+8+r.
#pragma unroll
    for (int r = 0; r < 8; ++r) {
      const int m = m0 + grp * 8 + r;
      float* orow = xg + (size_t)m * NH;
      orow[lo] = acc0[r] + bias0;
      if (lo < 8) orow[16 + lo] = acc1[r] + bias1;
    }
  }
}

// ---------------------------------------------------------------------------
// Phase 2: recurrence.  Each batch element is owned by a quad of lanes
// (lanes 4b..4b+3); lane q of the quad owns PyTorch gate group q (i,f,g,o),
// i.e. rows q*6..q*6+5 of W_hh.  Per step: 36 FMAs + 6 activations per lane,
// then ds_swizzle quad-broadcasts share the activated gates, and every lane
// redundantly updates c/h (needed for the next step's FMAs).
// ---------------------------------------------------------------------------
__device__ __forceinline__ float fast_sigmoid(float x) {
  return __builtin_amdgcn_rcpf(1.f + __expf(-x));
}
__device__ __forceinline__ float fast_tanh(float x) {
  x = fminf(10.f, fmaxf(-10.f, x));
  const float e = __expf(x + x);
  return (e - 1.f) * __builtin_amdgcn_rcpf(e + 1.f);
}

template <int S>
__device__ __forceinline__ float quad_bcast(float v) {
  // ds_swizzle group-of-4 mode: offset[15]=1, four 2-bit selects all = S.
  const int r = __builtin_amdgcn_ds_swizzle(__float_as_int(v),
                                            0x8000 | (S * 0x55));
  return __int_as_float(r);
}

__device__ __forceinline__ void load6(const float* __restrict__ p, float* dst) {
  const float2 a = *(const float2*)(p);       // 8-byte aligned (offsets % 8 == 0)
  const float2 b = *(const float2*)(p + 2);
  const float2 c = *(const float2*)(p + 4);
  dst[0] = a.x; dst[1] = a.y; dst[2] = b.x;
  dst[3] = b.y; dst[4] = c.x; dst[5] = c.y;
}

// buf[6] holds this lane's 6 gate pre-activations' xg part; consumed in place.
__device__ __forceinline__ void lstm_step_quad(const float* buf,
                                               const float W[HID][HID],
                                               bool isG,
                                               float* h, float* c, float* acc) {
  float g[HID];
#pragma unroll
  for (int k = 0; k < HID; ++k) g[k] = buf[k];
#pragma unroll
  for (int j = 0; j < HID; ++j) {
    const float hj = h[j];
#pragma unroll
    for (int k = 0; k < HID; ++k) g[k] = __builtin_fmaf(W[k][j], hj, g[k]);
  }
  float a[HID];
  if (isG) {          // lane owns the cell-candidate group -> tanh
#pragma unroll
    for (int k = 0; k < HID; ++k) a[k] = fast_tanh(g[k]);
  } else {            // i, f, o groups -> sigmoid
#pragma unroll
    for (int k = 0; k < HID; ++k) a[k] = fast_sigmoid(g[k]);
  }
  // Share activated gates across the quad (all lanes active here).
  float iv[HID], fv[HID], gv[HID], ov[HID];
#pragma unroll
  for (int k = 0; k < HID; ++k) {
    iv[k] = quad_bcast<0>(a[k]);
    fv[k] = quad_bcast<1>(a[k]);
    gv[k] = quad_bcast<2>(a[k]);
    ov[k] = quad_bcast<3>(a[k]);
  }
#pragma unroll
  for (int k = 0; k < HID; ++k) {
    const float cn = __builtin_fmaf(fv[k], c[k], iv[k] * gv[k]);
    c[k] = cn;
    const float hn = ov[k] * fast_tanh(cn);
    h[k] = hn;
    acc[k] += hn;
  }
}

__global__ __launch_bounds__(32) void lstm_recur(
    const float* __restrict__ xg, const float* __restrict__ W_hh,
    float* __restrict__ out) {
  const int lane = threadIdx.x;            // 0..31
  const int q    = lane & 3;               // gate group: 0=i, 1=f, 2=g, 3=o
  const int bi   = blockIdx.x * 8 + (lane >> 2);
  const bool isG = (q == 2);

  // This lane's 6x6 slice of W_hh (rows q*6 .. q*6+5).
  float W[HID][HID];
  const float* wrow = W_hh + q * HID * HID;
#pragma unroll
  for (int k = 0; k < HID; ++k)
#pragma unroll
    for (int j = 0; j < HID; ++j) W[k][j] = wrow[k * HID + j];

  float h[HID], c[HID], acc[HID];
#pragma unroll
  for (int k = 0; k < HID; ++k) { h[k] = 0.f; c[k] = 0.f; acc[k] = 0.f; }

  const float* gptr = xg + (size_t)bi * NH + q * HID;
  const size_t stride = (size_t)BATCH * NH;       // floats per timestep

  float bufA[HID], bufB[HID];
  load6(gptr, bufA);
  for (int t = 0; t < T_STEPS; t += 2) {
    load6(gptr + (size_t)(t + 1) * stride, bufB);
    lstm_step_quad(bufA, W, isG, h, c, acc);
    if (t + 2 < T_STEPS) load6(gptr + (size_t)(t + 2) * stride, bufA);
    lstm_step_quad(bufB, W, isG, h, c, acc);
  }

  if (q == 0) {
#pragma unroll
    for (int k = 0; k < HID; ++k) out[bi * HID + k] = acc[k];
  }
}

// ---------------------------------------------------------------------------
extern "C" void kernel_launch(void* const* d_in, const int* in_sizes, int n_in,
                              void* d_out, int out_size, void* d_ws, size_t ws_size,
                              hipStream_t stream) {
  (void)in_sizes; (void)n_in; (void)out_size; (void)ws_size;
  const float* x    = (const float*)d_in[0];   // [2048,256,100]
  const float* W_ih = (const float*)d_in[1];   // [24,100]
  const float* W_hh = (const float*)d_in[2];   // [24,6]
  const float* b_ih = (const float*)d_in[3];   // [24]
  const float* b_hh = (const float*)d_in[4];   // [24]
  float* out = (float*)d_out;                  // [256,6]
  float* xg  = (float*)d_ws;                   // [2048*256, 24] = 48 MiB scratch

  lstm_xproj_wmma<<<2048, 256, 0, stream>>>(x, W_ih, b_ih, b_hh, xg);
  lstm_recur<<<32, 32, 0, stream>>>(xg, W_hh, out);
}
